// InteractionModuleSkip_11347303596605
// MI455X (gfx1250) — matrix operand order repaired and checked by reading, not compiled, not run on hardware
//
#include <hip/hip_runtime.h>
#include <hip/hip_fp16.h>

typedef __attribute__((ext_vector_type(16))) _Float16 v16h;
typedef __attribute__((ext_vector_type(8)))  _Float16 v8h;
typedef __attribute__((ext_vector_type(8)))  float    v8f;

#define NVEC 1024
#define NDIM 64
#define NHID 32
#define NW   10
#define NCLS 10
#define NTHREADS 1024

// Branch-free tanh-GELU: gelu(x) = x * sigmoid(2*sqrt(2/pi)*(x + 0.044715 x^3)).
// Uses raw v_exp_f32 / v_rcp_f32 -> no EXEC divergence, ~7 VALU ops.
// Tails are exact: e->inf => x, e->0 => 0.
__device__ __forceinline__ float gelu_fast(float x) {
    const float u = x * (0.7978845608028654f + 0.0356774081363001f * x * x);
    // e = exp(2u) = 2^(2u*log2(e))
    const float e = __builtin_amdgcn_exp2f(u * 2.8853900817779268f);
    const float r = __builtin_amdgcn_rcpf(e + 1.0f);   // 1/(e+1) = 1 - sigmoid(2u)
    return x - x * r;                                   // x * sigmoid(2u)
}

// One workgroup per batch element. V (f32, column-major [64][1024]) and
// h (f16, [1024][32]) live in LDS: 256KB + 64KB = 320KB (full CDNA5 WGP LDS).
extern "C" __global__ __launch_bounds__(NTHREADS)
void chordnet_kernel(const float* __restrict__ data,
                     const float* __restrict__ fW1,
                     const float* __restrict__ fb1,
                     const float* __restrict__ fW2,
                     const float* __restrict__ fb2,
                     const float* __restrict__ finW,
                     const float* __restrict__ finb,
                     float* __restrict__ out) {
    extern __shared__ char smem[];
    float*    Vs = (float*)smem;                             // [64][1024] f32, 256KB
    _Float16* hs = (_Float16*)(smem + NVEC * NDIM * 4);      // [1024][32] f16, 64KB

    const int b    = blockIdx.x;
    const int tid  = threadIdx.x;
    const int lane = tid & 31;
    const int wave = tid >> 5;
    const int lo16  = lane & 15;
    const int khalf = lane >> 4;

    const float* dataB = data + (size_t)b * NVEC * NDIM;

    // ---- init: V = data (transpose into column-major LDS) ----
    for (int idx = tid; idx < NVEC * NDIM; idx += NTHREADS) {
        int n = idx >> 6, d = idx & 63;
        Vs[d * NVEC + n] = dataB[idx];
    }
    __syncthreads();

    const int off[11] = {0, 1, 2, 4, 8, 16, 32, 64, 128, 256, 512};

    for (int layer = NW - 1; layer >= 0; --layer) {
        const float* W1 = fW1 + layer * NDIM * NHID;   // [64][32]
        const float* b1 = fb1 + layer * NHID;
        const float* W2 = fW2 + layer * NHID * NVEC;   // [32][1024]
        const float* b2 = fb2 + layer * NVEC;

        // ================= Phase 1: h = gelu(data @ W1 + b1) via WMMA =================
        // B operand tiles (32x16 f16): lane<16 holds K={0..7}U{16..23} of column lo16,
        // lane>=16 holds K={8..15}U{24..31} of column lo16.
        v16h Bt[2][2];
        #pragma unroll
        for (int nt = 0; nt < 2; ++nt) {
            #pragma unroll
            for (int kk = 0; kk < 2; ++kk) {
                const int col = nt * 16 + lo16;
                const int kb  = kk * 32 + 8 * khalf;
                v16h bv;
                #pragma unroll
                for (int e = 0; e < 16; ++e)
                    bv[e] = (_Float16)W1[(kb + e + (e >= 8 ? 8 : 0)) * NHID + col];
                Bt[nt][kk] = bv;
            }
        }
        const float bias0 = b1[lo16];        // column bias, lane-uniform over rows
        const float bias1 = b1[16 + lo16];
        #pragma unroll
        for (int mi = 0; mi < 2; ++mi) {
            const int mtile = wave * 2 + mi;          // 32 waves x 2 = 64 M-tiles
            const int mbase = mtile * 16;
            const int arow  = mbase + lo16;
            // A operand (16x32 f16): lane<16 row=arow K={0..7}U{16..23}(+kbase);
            // lane>=16 K={8..15}U{24..31}(+kbase) -> two contiguous runs of 8 floats.
            v16h At[2];
            #pragma unroll
            for (int kk = 0; kk < 2; ++kk) {
                const int k0 = kk * 32 + 8 * khalf;
                const int k1 = kk * 32 + 16 + 8 * khalf;
                const float4* r0 = (const float4*)(dataB + arow * NDIM + k0);
                const float4* r1 = (const float4*)(dataB + arow * NDIM + k1);
                float4 p0 = r0[0], p1 = r0[1], q0 = r1[0], q1 = r1[1];
                v16h av;
                av[0]  = (_Float16)p0.x; av[1]  = (_Float16)p0.y;
                av[2]  = (_Float16)p0.z; av[3]  = (_Float16)p0.w;
                av[4]  = (_Float16)p1.x; av[5]  = (_Float16)p1.y;
                av[6]  = (_Float16)p1.z; av[7]  = (_Float16)p1.w;
                av[8]  = (_Float16)q0.x; av[9]  = (_Float16)q0.y;
                av[10] = (_Float16)q0.z; av[11] = (_Float16)q0.w;
                av[12] = (_Float16)q1.x; av[13] = (_Float16)q1.y;
                av[14] = (_Float16)q1.z; av[15] = (_Float16)q1.w;
                At[kk] = av;
            }
            #pragma unroll
            for (int nt = 0; nt < 2; ++nt) {
                // Seed accumulator with the bias (C layout: one column per lane,
                // so the bias is identical across all 8 C VGPRs of a lane).
                const float bias = nt ? bias1 : bias0;
                v8f c;
                #pragma unroll
                for (int r = 0; r < 8; ++r) c[r] = bias;
                c = __builtin_amdgcn_wmma_f32_16x16x32_f16(
                        false, At[0], false, Bt[nt][0], (short)0, c, false, false);
                c = __builtin_amdgcn_wmma_f32_16x16x32_f16(
                        false, At[1], false, Bt[nt][1], (short)0, c, false, false);
                const int col = nt * 16 + lo16;
                // C/D layout: VGPR r, lanes<16 -> M=r, lanes>=16 -> M=r+8, N=lane%16
                #pragma unroll
                for (int r = 0; r < 8; ++r) {
                    const int m = mbase + r + 8 * khalf;
                    hs[m * NHID + col] = (_Float16)gelu_fast(c[r]);
                }
            }
        }
        __syncthreads();

        // ================= Phase 2: sparse V update (11 taps/row) =================
        const int n = tid;  // one row per thread
        float coef[11];
        int   mk[11];
        {
            // vectorized f16 row load: 4 x 16B
            float hrow[NHID];
            const v8h* h8 = (const v8h*)(hs + n * NHID);
            #pragma unroll
            for (int c4 = 0; c4 < 4; ++c4) {
                v8h hv = h8[c4];
                #pragma unroll
                for (int e = 0; e < 8; ++e) hrow[c4 * 8 + e] = (float)hv[e];
            }
            #pragma unroll
            for (int k = 0; k < 11; ++k) {
                const int m = (n + off[k]) & (NVEC - 1);
                mk[k] = m;
                float cf = b2[m];
                #pragma unroll
                for (int j = 0; j < NHID; ++j)
                    cf += hrow[j] * W2[j * NVEC + m];   // coalesced across lanes
                coef[k] = cf;
            }
        }
        // Column-independent update: process d in 2 chunks of 32 to keep the
        // accumulator footprint at 32 VGPRs (no spills under 1024-thread bounds).
        #pragma unroll
        for (int chunk = 0; chunk < 2; ++chunk) {
            const int dbase = chunk * 32;
            float acc[32];
            #pragma unroll
            for (int d = 0; d < 32; ++d)        // residual (conflict-free: lanes walk n)
                acc[d] = Vs[(dbase + d) * NVEC + n];
            #pragma unroll
            for (int k = 0; k < 11; ++k) {
                const float cf = coef[k];
                const int   m  = mk[k];
                #pragma unroll
                for (int d = 0; d < 32; ++d)
                    acc[d] += cf * Vs[(dbase + d) * NVEC + m];
            }
            __syncthreads();   // all reads of these columns complete
            #pragma unroll
            for (int d = 0; d < 32; ++d)
                Vs[(dbase + d) * NVEC + n] = acc[d];
            __syncthreads();   // new columns visible
        }
    }

    // ================= Phase 3: out = flat(V) @ finW + finb =================
    float part[NCLS];
    #pragma unroll
    for (int c = 0; c < NCLS; ++c) part[c] = 0.0f;
    const int n = tid;  // one row (64 elems) per thread; flat idx e = n*64+d
    for (int d = 0; d < NDIM; ++d) {
        const float v = Vs[d * NVEC + n];
        const float* w = finW + (size_t)(n * NDIM + d) * NCLS;
        #pragma unroll
        for (int c = 0; c < NCLS; ++c) part[c] += v * w[c];
    }
    #pragma unroll
    for (int c = 0; c < NCLS; ++c) {
        #pragma unroll
        for (int s = 16; s >= 1; s >>= 1)
            part[c] += __shfl_xor(part[c], s, 32);   // wave32 reduction
    }
    float* red = (float*)hs;   // reuse h region as reduction scratch
    if (lane == 0) {
        #pragma unroll
        for (int c = 0; c < NCLS; ++c) red[wave * NCLS + c] = part[c];
    }
    __syncthreads();
    if (tid < NCLS) {
        float s = finb[tid];
        for (int w = 0; w < NTHREADS / 32; ++w) s += red[w * NCLS + tid];
        out[b * NCLS + tid] = s;
    }
}

extern "C" void kernel_launch(void* const* d_in, const int* in_sizes, int n_in,
                              void* d_out, int out_size, void* d_ws, size_t ws_size,
                              hipStream_t stream) {
    const float* data = (const float*)d_in[0];
    const float* fW1  = (const float*)d_in[1];
    const float* fb1  = (const float*)d_in[2];
    const float* fW2  = (const float*)d_in[3];
    const float* fb2  = (const float*)d_in[4];
    const float* finW = (const float*)d_in[5];
    const float* finb = (const float*)d_in[6];
    float* out = (float*)d_out;

    const int B = in_sizes[0] / (NVEC * NDIM);   // 16
    const size_t smem = (size_t)NVEC * NDIM * sizeof(float)
                      + (size_t)NVEC * NHID * sizeof(_Float16);  // 320 KB

    chordnet_kernel<<<dim3(B), dim3(NTHREADS), smem, stream>>>(
        data, fW1, fb1, fW2, fb2, finW, finb, out);
}
